// LowRankSlotElmanCell_24404004176350
// MI455X (gfx1250) — compile-verified, matrix operand order checked by reading, not run on hardware
//
#include <hip/hip_runtime.h>
#include <hip/hip_bf16.h>
#include <math.h>

#define DIM     1024
#define N_SLOTS 8
#define RANK    256
#define T_STEPS 1024
#define BATCH   8

typedef _Float16 v16h __attribute__((ext_vector_type(16)));
typedef float    v8f  __attribute__((ext_vector_type(8)));

static __device__ __forceinline__ v8f wmma_f16f32(v16h a, v16h b, v8f c) {
    // D = A(16x32 f16) * B(32x16 f16) + C(16x16 f32)
    return __builtin_amdgcn_wmma_f32_16x16x32_f16(
        /*neg_a=*/false, a, /*neg_b=*/false, b,
        /*c_mod=*/(short)0, c, /*reuse_a=*/false, /*reuse_b=*/false);
}

// Load 16 contiguous f32 and convert to an f16 fragment run.
static __device__ __forceinline__ v16h frag_from_f32(const float* __restrict__ p) {
    v16h r;
#pragma unroll
    for (int i = 0; i < 16; ++i) r[i] = (_Float16)p[i];
    return r;
}

// ---------------------------------------------------------------------------
// Kernel 1: f32 -> f16 convert (for U and V weight tensors)
// ---------------------------------------------------------------------------
__global__ void cvt_f32_f16_kernel(const float* __restrict__ src,
                                   _Float16* __restrict__ dst, int n) {
    int i = blockIdx.x * blockDim.x + threadIdx.x;
    if (i < n) dst[i] = (_Float16)src[i];
}

// ---------------------------------------------------------------------------
// Kernel 2: Wx = x @ W_x^T + b   (M = T*B = 8192, N = DIM, K = DIM)
// One 16x16 output tile per wave, 8 waves / 256-thread block.
// ---------------------------------------------------------------------------
__global__ void wx_gemm_kernel(const float* __restrict__ x,
                               const float* __restrict__ Wm,   // [DIM out][DIM in] row-major
                               const float* __restrict__ bias,
                               float* __restrict__ wx) {
    const int lane = threadIdx.x & 31;
    const int wave = threadIdx.x >> 5;
    const int tile = blockIdx.x * 8 + wave;       // 32768 tiles total
    const int mt = tile >> 6;                     // 512 M-tiles
    const int nt = tile & 63;                     // 64  N-tiles
    const int row0 = mt * 16, col0 = nt * 16;
    const int mn   = lane & 15;                   // row (A) == col (B) index
    const int ksel = lane >> 4;                   // which 16-wide K half

    const float* __restrict__ arow = x  + (size_t)(row0 + mn) * DIM + ksel * 16;
    const float* __restrict__ brow = Wm + (size_t)(col0 + mn) * DIM + ksel * 16; // B[k][n]=W[n][k]

    // dual accumulators: two independent WMMA chains
    v8f c0 = {}, c1 = {};
#pragma unroll 2
    for (int k = 0; k < DIM; k += 64) {
        c0 = wmma_f16f32(frag_from_f32(arow + k),      frag_from_f32(brow + k),      c0);
        c1 = wmma_f16f32(frag_from_f32(arow + k + 32), frag_from_f32(brow + k + 32), c1);
    }
#pragma unroll
    for (int i = 0; i < 8; ++i) {
        const int r = row0 + i + 8 * ksel;
        const int d = col0 + mn;
        wx[(size_t)r * DIM + d] = (c0[i] + c1[i]) + bias[d];
    }
}

// ---------------------------------------------------------------------------
// Kernel 3: write h[0, s, b, :] = h0[b, s, :]
// ---------------------------------------------------------------------------
__global__ void h0_copy_kernel(const float* __restrict__ h0,
                               float* __restrict__ h_out) {
    const int idx = blockIdx.x * blockDim.x + threadIdx.x;  // over S*B*DIM
    if (idx >= N_SLOTS * BATCH * DIM) return;
    const int d = idx % DIM;
    const int r = idx / DIM;
    const int b = r % BATCH;
    const int s = r / BATCH;
    h_out[idx] = h0[((size_t)b * N_SLOTS + s) * DIM + d];   // idx == (s*B+b)*DIM+d
}

// ---------------------------------------------------------------------------
// Kernel 4: the recurrence. One workgroup per slot (8 blocks x 512 threads
// = 16 wave32s). h state lives in LDS as f16 (16 rows: 8 batch + 8 pad).
//   GEMM1: Vh(16x256) = h(16x1024) @ V[s]^T   (1 N-tile / wave, V streamed,
//                                              dual-accumulator WMMA chains)
//   GEMM2: Uh(16x1024) = Vh(16x256) @ U[s]^T  (4 N-tiles / wave; U held in a
//                                              256-VGPR register cache; kk-outer
//                                              loop -> 4 independent WMMA chains
//                                              and 4x A-fragment reuse from LDS)
//   h_new = tanh(Uh + Wx[t]) -> global h_out (non-temporal) + back into LDS
// ---------------------------------------------------------------------------
__global__ void __launch_bounds__(512, 1)
recurrence_kernel(const float* __restrict__ h0,
                  const float* __restrict__ wx,      // [T*B, DIM] f32
                  const _Float16* __restrict__ V16,  // [S, RANK, DIM]
                  const _Float16* __restrict__ U16,  // [S, DIM, RANK]
                  float* __restrict__ h_out) {       // [T+1, S, B, DIM]
    __shared__ __attribute__((aligned(32))) _Float16 sh_h [16][DIM];
    __shared__ __attribute__((aligned(32))) _Float16 sh_vh[16][RANK];

    const int s    = blockIdx.x;          // slot
    const int lane = threadIdx.x & 31;
    const int wave = threadIdx.x >> 5;    // 0..15
    const int mn   = lane & 15;
    const int ksel = lane >> 4;

    // init LDS state from h0 (pad rows 8..15 = 0)
    for (int idx = threadIdx.x; idx < 16 * DIM; idx += 512) {
        const int mr = idx / DIM, d = idx % DIM;
        const float v = (mr < BATCH)
            ? h0[((size_t)mr * N_SLOTS + s) * DIM + d] : 0.f;
        sh_h[mr][d] = (_Float16)v;
    }

    const _Float16* __restrict__ Vs = V16 + (size_t)s * RANK * DIM;
    const _Float16* __restrict__ Us = U16 + (size_t)s * DIM * RANK;

    // ---- register-cache this wave's GEMM2 B fragments (loop-invariant) ----
    // 4 tiles x 8 k-steps x v16h = 32 v16h = 256 VGPRs/lane.
    v16h bU[4][8];
#pragma unroll
    for (int tt = 0; tt < 4; ++tt) {
        const int col0 = (wave * 4 + tt) * 16;
        const _Float16* __restrict__ br =
            Us + (size_t)(col0 + mn) * RANK + ksel * 16;      // B[k][n] = U[n][k]
#pragma unroll
        for (int kk = 0; kk < 8; ++kk)
            bU[tt][kk] = *(const v16h*)(br + kk * 32);
    }
    __syncthreads();

    const int g1col0 = wave * 16;   // GEMM1 N-tile for this wave
    const _Float16* __restrict__ vrow =
        Vs + (size_t)(g1col0 + mn) * DIM + ksel * 16;         // B[k][n] = V[n][k]

    for (int t = 0; t < T_STEPS; ++t) {
        // pull next step's 32 KB Wx block toward the WGP while we compute
        if (t + 1 < T_STEPS) {
            __builtin_prefetch(wx + (size_t)(t + 1) * BATCH * DIM
                                  + (size_t)threadIdx.x * 16, 0, 1);
        }

        // ---- GEMM1: Vh = h @ V^T ; wave owns cols [wave*16, wave*16+16) ----
        {
            const _Float16* __restrict__ arow = &sh_h[mn][ksel * 16];
            v8f c0 = {}, c1 = {};   // two independent accumulation chains
#pragma unroll 2
            for (int k = 0; k < DIM; k += 64) {
                v16h a0 = *(const v16h*)(arow + k);
                v16h b0 = *(const v16h*)(vrow + k);
                v16h a1 = *(const v16h*)(arow + k + 32);
                v16h b1 = *(const v16h*)(vrow + k + 32);
                c0 = wmma_f16f32(a0, b0, c0);
                c1 = wmma_f16f32(a1, b1, c1);
            }
#pragma unroll
            for (int i = 0; i < 8; ++i)
                sh_vh[i + 8 * ksel][g1col0 + mn] = (_Float16)(c0[i] + c1[i]);
        }
        __syncthreads();   // Vh complete; h reads of this step done

        // ---- GEMM2: Uh = Vh @ U^T ; kk-outer: 4 independent chains,
        //      each LDS A-fragment read once and used for 4 WMMAs ----
        const float* __restrict__ wx_t = wx + (size_t)t * BATCH * DIM;
        const _Float16* __restrict__ arow2 = &sh_vh[mn][ksel * 16];
        v8f c[4] = {{}, {}, {}, {}};
#pragma unroll
        for (int kk = 0; kk < 8; ++kk) {
            v16h a = *(const v16h*)(arow2 + kk * 32);
#pragma unroll
            for (int tt = 0; tt < 4; ++tt)
                c[tt] = wmma_f16f32(a, bU[tt][kk], c[tt]);
        }
#pragma unroll
        for (int tt = 0; tt < 4; ++tt) {
            const int col0 = (wave * 4 + tt) * 16;
#pragma unroll
            for (int i = 0; i < 8; ++i) {
                const int mr = i + 8 * ksel;
                const int d  = col0 + mn;
                const float val =
                    tanhf(c[tt][i] + wx_t[(size_t)(mr & 7) * DIM + d]);
                if (mr < BATCH) {
                    // streaming store: don't let h evict V/U from L2
                    __builtin_nontemporal_store(
                        val,
                        &h_out[((((size_t)(t + 1) * N_SLOTS + s) * BATCH) + mr)
                                   * DIM + d]);
                }
                sh_h[mr][d] = (_Float16)val;
            }
        }
        __syncthreads();   // h updated before next step's GEMM1 reads
    }
}

// ---------------------------------------------------------------------------
// Kernel 5: out[t,b,d] = (sum_s C[s]*h[t+1,s,b,d]) * silu(z[t,b,d])
// ---------------------------------------------------------------------------
__global__ void output_kernel(const float* __restrict__ z,
                              const float* __restrict__ C,
                              const float* __restrict__ h_all,
                              float* __restrict__ out) {
    const size_t idx = (size_t)blockIdx.x * blockDim.x + threadIdx.x;
    if (idx >= (size_t)T_STEPS * BATCH * DIM) return;
    const int d = (int)(idx % DIM);
    const size_t tb = idx / DIM;
    const int b = (int)(tb % BATCH);
    const size_t t = tb / BATCH;
    float acc = 0.f;
#pragma unroll
    for (int s = 0; s < N_SLOTS; ++s)
        acc += C[s] * h_all[(((t + 1) * N_SLOTS + s) * BATCH + b) * DIM + d];
    const float zz = z[idx];
    __builtin_nontemporal_store(acc * (zz / (1.f + expf(-zz))), &out[idx]);
}

// ---------------------------------------------------------------------------
extern "C" void kernel_launch(void* const* d_in, const int* in_sizes, int n_in,
                              void* d_out, int out_size, void* d_ws, size_t ws_size,
                              hipStream_t stream) {
    const float* x   = (const float*)d_in[0];  // [T,B,D]
    const float* z   = (const float*)d_in[1];  // [T,B,D]
    const float* h0  = (const float*)d_in[2];  // [B,S,D]
    const float* W_x = (const float*)d_in[3];  // [D,D]
    const float* U   = (const float*)d_in[4];  // [S,D,R]
    const float* V   = (const float*)d_in[5];  // [S,R,D]
    const float* bia = (const float*)d_in[6];  // [D]
    const float* C   = (const float*)d_in[7];  // [S]

    float* out   = (float*)d_out;                                 // [T,B,D]
    float* h_all = (float*)d_out + (size_t)T_STEPS * BATCH * DIM; // [T+1,S,B,D]

    // workspace layout
    char* ws = (char*)d_ws;
    float*    wx_buf = (float*)ws;                                          // 32 MB
    _Float16* V16    = (_Float16*)(ws + (size_t)T_STEPS * BATCH * DIM * 4); // 4 MB
    _Float16* U16    = V16 + (size_t)N_SLOTS * RANK * DIM;                  // 4 MB

    const int nW = N_SLOTS * RANK * DIM;  // == N_SLOTS*DIM*RANK
    cvt_f32_f16_kernel<<<(nW + 255) / 256, 256, 0, stream>>>(V, V16, nW);
    cvt_f32_f16_kernel<<<(nW + 255) / 256, 256, 0, stream>>>(U, U16, nW);

    // Wx = x @ W_x^T + b : 512 x 64 tiles, 8 tiles (waves) per block
    wx_gemm_kernel<<<4096, 256, 0, stream>>>(x, W_x, bia, wx_buf);

    h0_copy_kernel<<<(N_SLOTS * BATCH * DIM + 255) / 256, 256, 0, stream>>>(h0, h_all);

    recurrence_kernel<<<N_SLOTS, 512, 0, stream>>>(h0, wx_buf, V16, U16, h_all);

    const size_t nOut = (size_t)T_STEPS * BATCH * DIM;
    output_kernel<<<(unsigned)((nOut + 255) / 256), 256, 0, stream>>>(z, C, h_all, out);
}
